// LineRWKVPredictor_56564719288737
// MI455X (gfx1250) — matrix operand order repaired and checked by reading, not compiled, not run on hardware
//
#include <hip/hip_runtime.h>

// ---------------------------------------------------------------------------
// LineRWKV predictor for gfx1250 (MI455X).
// GEMMs on v_wmma_f32_16x16x32_bf16; weights staged in LDS via the Tensor
// Data Mover (tensor_load_to_lds, TENSORcnt) with async-to-LDS fallback;
// one workgroup per batch element keeps the [512][128] line resident in LDS.
// 512 threads/block (16 waves, 2 M-tiles per wave) => ~256 VGPR budget/wave.
// ---------------------------------------------------------------------------

typedef __attribute__((ext_vector_type(16))) __bf16 v16bf;
typedef __attribute__((ext_vector_type(8)))  float  v8f;
typedef __attribute__((ext_vector_type(4)))  unsigned int u32x4;
typedef __attribute__((ext_vector_type(8)))  int    i32x8;
typedef __attribute__((ext_vector_type(4)))  int    i32x4;
typedef __attribute__((ext_vector_type(4)))  int    v4i;
typedef __attribute__((ext_vector_type(2)))  float  f32x2;

union FragB16 { v16bf v; u32x4 u[2]; };

#define B_   2
#define H_   128
#define W_   512
#define C_   202
#define D_   128
#define L_   4
#define CP_IN  224   // C padded to multiple of 32 (input proj K)
#define CP_OUT 208   // C padded to multiple of 16 (output proj N)
#define NTHR 512     // threads per pipeline block (16 waves)

// ----- LDS byte offsets (302 KB total of the 320 KB WGP pool) --------------
#define LS_LINE   0          // [512][128] bf16 : 131072
#define LS_STATE  131072     // [L][128] f32    : 2048
#define LS_WIN    133120     // [128][224] bf16 : 57344
#define LS_WOUT   190464     // [208][128] bf16 : 53248
#define LS_WV     243712     // [128][128] bf16 : 32768 (reused for Wcm)
#define LS_WR     276480     // [128][128] bf16 : 32768
#define LS_TOTAL  309248

#if defined(__AMDGCN__) && __has_builtin(__builtin_amdgcn_global_load_async_to_lds_b128)
#define HAS_ASYNC_LDS 1
#else
#define HAS_ASYNC_LDS 0
#endif

#if defined(__AMDGCN__) && __has_builtin(__builtin_amdgcn_tensor_load_to_lds)
#define HAS_TDM 1
#else
#define HAS_TDM 0
#endif

__device__ __forceinline__ void copy16_g2l(const void* g, void* l) {
#if HAS_ASYNC_LDS
  __builtin_amdgcn_global_load_async_to_lds_b128(
      (__attribute__((address_space(1))) v4i*)g,
      (__attribute__((address_space(3))) v4i*)l, 0, 0);
#else
  *(u32x4*)l = *(const u32x4*)g;
#endif
}

__device__ __forceinline__ void wait_async() {
#if HAS_ASYNC_LDS
#if __has_builtin(__builtin_amdgcn_s_wait_asynccnt)
  __builtin_amdgcn_s_wait_asynccnt(0);
#else
  asm volatile("s_wait_asynccnt 0" ::: "memory");
#endif
#endif
}

// Copy `bytes` (multiple of 16) global->LDS, strided over `nthreads` threads.
__device__ __forceinline__ void stage_g2l(const void* g, void* l, int bytes,
                                          int tid, int nthreads) {
  const char* gp = (const char*)g;
  char* lp = (char*)l;
  for (int off = tid * 16; off < bytes; off += nthreads * 16)
    copy16_g2l(gp + off, lp + off);
}

#if HAS_TDM
__device__ __forceinline__ unsigned lds_addr_of(const void* p) {
  return (unsigned)(unsigned long long)(__attribute__((address_space(3))) const char*)p;
}

// TDM 1-D tile load: nelem bf16 elements, contiguous global -> LDS.
// D# per CDNA5 ISA 8.3/8.4: count=1, type=2, data_size=2B,
// tensor_dim0 = tile_dim0 = stride0 = nelem (< 65536), dims 1+ unused.
__device__ __forceinline__ void tdm_load_bf16(const void* g, void* l,
                                              unsigned nelem) {
  const unsigned long long ga = (unsigned long long)g;
  u32x4 g0;
  g0[0] = 1u;                                           // count=1 (user mode)
  g0[1] = lds_addr_of(l);                               // lds_addr
  g0[2] = (unsigned)(ga & 0xffffffffu);                 // global_addr[31:0]
  g0[3] = (unsigned)((ga >> 32) & 0x1ffffffu) | (2u << 30);  // [56:32] | type=2
  i32x8 g1;
  g1[0] = (int)(1u << 16);        // data_size=1 (2 bytes), mask/flags 0
  g1[1] = (int)(nelem << 16);     // tensor_dim0[15:0] @ bits 63:48
  g1[2] = (int)(1u << 16);        // tensor_dim0[31:16]=0, tensor_dim1=1
  g1[3] = (int)(nelem << 16);     // tensor_dim1[31:16]=0, tile_dim0
  g1[4] = 0;                      // tile_dim1=0 (unused), tile_dim2=0
  g1[5] = (int)nelem;             // tensor_dim0_stride[31:0]
  g1[6] = 0;                      // stride0[47:32]=0, stride1[15:0]=0
  g1[7] = 0;                      // stride1[47:16]=0
  i32x4 z4 = {0, 0, 0, 0};
#if __clang_major__ >= 23
  i32x8 z8 = {0, 0, 0, 0, 0, 0, 0, 0};
  __builtin_amdgcn_tensor_load_to_lds(g0, g1, z4, z4, z8, 0);
#else
  __builtin_amdgcn_tensor_load_to_lds(g0, g1, z4, z4, 0);
#endif
}

__device__ __forceinline__ void wait_tensor() {
#if __has_builtin(__builtin_amdgcn_s_wait_tensorcnt)
  __builtin_amdgcn_s_wait_tensorcnt(0);
#else
  asm volatile("s_wait_tensorcnt 0" ::: "memory");
#endif
}
#endif  // HAS_TDM

__device__ __forceinline__ v8f wmma_bf16(v16bf a, v16bf b, v8f c) {
  return __builtin_amdgcn_wmma_f32_16x16x32_bf16(
      false, a, false, b, (short)0, c, false, false);
}

__device__ __forceinline__ v8f splat8(float x) {
  v8f r;
  #pragma unroll
  for (int i = 0; i < 8; ++i) r[i] = x;
  return r;
}

// A fragment: 16x32 bf16 tile, row-major, row stride ld.
__device__ __forceinline__ v16bf load_a_bf16(const __bf16* src, int ld, int lane) {
  const int m  = lane & 15;
  const int k0 = (lane < 16) ? 0 : 8;
  FragB16 f;
  const __bf16* p = src + (size_t)m * ld + k0;
  f.u[0] = *(const u32x4*)(p);
  f.u[1] = *(const u32x4*)(p + 16);
  return f.v;
}

// B fragment: 32x16 tile from transposed weights Wt[n][k]; k = kb..kb+7, kb+16..kb+23 per lane.
__device__ __forceinline__ v16bf load_b_bf16(const __bf16* wt, int ld, int lane) {
  const int n  = lane & 15;
  const int kb = (lane < 16) ? 0 : 8;
  FragB16 f;
  const __bf16* p = wt + (size_t)n * ld + kb;
  f.u[0] = *(const u32x4*)(p);
  f.u[1] = *(const u32x4*)(p + 16);
  return f.v;
}

// ---------------------------------------------------------------------------
// Weight prep: transpose + bf16 + zero-pad (once per launch, deterministic).
// ---------------------------------------------------------------------------
__global__ void convert_weights(const float* __restrict__ w_in,
                                const float* __restrict__ w_v,
                                const float* __restrict__ w_r,
                                const float* __restrict__ w_cm,
                                const float* __restrict__ w_out,
                                __bf16* __restrict__ win_t,
                                __bf16* __restrict__ wv_t,
                                __bf16* __restrict__ wr_t,
                                __bf16* __restrict__ wcm_t,
                                __bf16* __restrict__ wout_t) {
  const int N_WIN  = D_ * CP_IN;
  const int N_MAT  = 3 * L_ * D_ * D_;
  const int N_WOUT = CP_OUT * D_;
  int idx = blockIdx.x * blockDim.x + threadIdx.x;
  if (idx < N_WIN) {
    int d = idx / CP_IN, cp = idx % CP_IN;
    win_t[idx] = (cp < C_) ? (__bf16)w_in[(size_t)cp * D_ + d] : (__bf16)0.f;
  } else if (idx < N_WIN + N_MAT) {
    int j   = idx - N_WIN;
    int mat = j / (L_ * D_ * D_);
    int e   = j % (L_ * D_ * D_);
    int l   = e / (D_ * D_);
    int r2  = e % (D_ * D_);
    int n   = r2 / D_, k = r2 % D_;
    const float* src = (mat == 0) ? w_v : (mat == 1) ? w_r : w_cm;
    __bf16*      dst = (mat == 0) ? wv_t : (mat == 1) ? wr_t : wcm_t;
    dst[(size_t)l * D_ * D_ + (size_t)n * D_ + k] =
        (__bf16)src[(size_t)l * D_ * D_ + (size_t)k * D_ + n];
  } else if (idx < N_WIN + N_MAT + N_WOUT) {
    int j = idx - N_WIN - N_MAT;
    int cp = j / D_, k = j % D_;
    wout_t[j] = (cp < C_) ? (__bf16)w_out[(size_t)k * C_ + cp] : (__bf16)0.f;
  }
}

// ---------------------------------------------------------------------------
// Sequential pipeline: 1 block per batch element, 512 threads = 16 waves.
// Each wave owns two 16-row M-tiles of the 512-row line.
// ---------------------------------------------------------------------------
__global__ __launch_bounds__(NTHR) void rwkv_pipeline(
    const float* __restrict__ x, const float* __restrict__ b_out,
    const __bf16* __restrict__ win_t, const __bf16* __restrict__ wv_t,
    const __bf16* __restrict__ wr_t, const __bf16* __restrict__ wcm_t,
    const __bf16* __restrict__ wout_t,
    float* __restrict__ ws_v, float* __restrict__ ws_r,
    float* __restrict__ out) {
  extern __shared__ char smem[];
  __bf16* ls_line  = (__bf16*)(smem + LS_LINE);
  float*  ls_state = (float*)(smem + LS_STATE);
  __bf16* ls_win   = (__bf16*)(smem + LS_WIN);
  __bf16* ls_wout  = (__bf16*)(smem + LS_WOUT);
  __bf16* ls_wv    = (__bf16*)(smem + LS_WV);   // Wv, later reused for Wcm
  __bf16* ls_wr    = (__bf16*)(smem + LS_WR);

  const int b     = blockIdx.x;
  const int tid   = threadIdx.x;
  const int wave  = tid >> 5;     // 0..15
  const int lane  = tid & 31;
  const int lhalf = lane >> 4;
  const int lmod  = lane & 15;

  float* wsv = ws_v + (size_t)b * W_ * D_;
  float* wsr = ws_r + (size_t)b * W_ * D_;

  // one-time staging of projection weights + state init
#if HAS_TDM
  if (wave == 0) {
    tdm_load_bf16(win_t,  ls_win,  D_ * CP_IN);
    tdm_load_bf16(wout_t, ls_wout, CP_OUT * D_);
    wait_tensor();
  }
#else
  stage_g2l(win_t,  ls_win,  D_ * CP_IN * 2,  tid, NTHR);
  stage_g2l(wout_t, ls_wout, CP_OUT * D_ * 2, tid, NTHR);
  wait_async();
#endif
  for (int i = tid; i < L_ * D_; i += NTHR) ls_state[i] = 0.f;
  __syncthreads();

  for (int h = 0; h < H_; ++h) {
    // ---- input projection: line = x[b,h] @ W_in  (K=202 pad 224) ----------
    for (int mh = 0; mh < 2; ++mh) {
      const int mb = (wave + 16 * mh) * 16;
      const float* xrow = x + ((((size_t)b * H_ + h) * W_) + (mb + lmod)) * C_;
      v8f acc[8];
      #pragma unroll
      for (int nt = 0; nt < 8; ++nt) acc[nt] = splat8(0.f);
      for (int kc = 0; kc < 7; ++kc) {
        const int k0 = kc * 32 + ((lane < 16) ? 0 : 8);
        v16bf a;
        if (kc < 6) {  // fully in-bounds: float2 vector loads (8B aligned)
          #pragma unroll
          for (int q = 0; q < 4; ++q) {
            f32x2 t1 = *(const f32x2*)(xrow + k0 + 2 * q);
            f32x2 t2 = *(const f32x2*)(xrow + k0 + 16 + 2 * q);
            a[2 * q]         = (__bf16)t1.x;
            a[2 * q + 1]     = (__bf16)t1.y;
            a[8 + 2 * q]     = (__bf16)t2.x;
            a[8 + 2 * q + 1] = (__bf16)t2.y;
          }
        } else {       // tail chunk: mask k >= 202
          #pragma unroll
          for (int e = 0; e < 8; ++e) {
            const int k1 = k0 + e, k2 = k0 + 16 + e;
            a[e]     = (k1 < C_) ? (__bf16)xrow[k1] : (__bf16)0.f;
            a[8 + e] = (k2 < C_) ? (__bf16)xrow[k2] : (__bf16)0.f;
          }
        }
        #pragma unroll
        for (int nt = 0; nt < 8; ++nt) {
          v16bf bw = load_b_bf16(ls_win + (size_t)nt * 16 * CP_IN + kc * 32,
                                 CP_IN, lane);
          acc[nt] = wmma_bf16(a, bw, acc[nt]);
        }
      }
      #pragma unroll
      for (int nt = 0; nt < 8; ++nt) {
        const int n = nt * 16 + lmod;
        #pragma unroll
        for (int v = 0; v < 8; ++v)
          ls_line[(size_t)(mb + v + lhalf * 8) * D_ + n] = (__bf16)acc[nt][v];
      }
    }

    // ---- layers (sequential) ----------------------------------------------
    for (int l = 0; l < L_; ++l) {
      __syncthreads();  // protect ls_wv/ls_wr restaging vs previous readers
#if HAS_TDM
      if (wave == 0) {
        tdm_load_bf16(wv_t + (size_t)l * D_ * D_, ls_wv, D_ * D_);
        tdm_load_bf16(wr_t + (size_t)l * D_ * D_, ls_wr, D_ * D_);
        wait_tensor();
      }
#else
      stage_g2l(wv_t + (size_t)l * D_ * D_, ls_wv, D_ * D_ * 2, tid, NTHR);
      stage_g2l(wr_t + (size_t)l * D_ * D_, ls_wr, D_ * D_ * 2, tid, NTHR);
      wait_async();
#endif
      __syncthreads();

      // v = line @ Wv ; r = sigmoid(line @ Wr).  kc-outer register blocking:
      // one A fragment per K-chunk feeds 16 WMMAs (accv[8] + accr[8]).
      for (int mh = 0; mh < 2; ++mh) {
        const int mb = (wave + 16 * mh) * 16;
        v8f accv[8], accr[8];
        #pragma unroll
        for (int nt = 0; nt < 8; ++nt) { accv[nt] = splat8(0.f); accr[nt] = splat8(0.f); }
        #pragma unroll
        for (int kc = 0; kc < 4; ++kc) {
          v16bf a = load_a_bf16(ls_line + (size_t)mb * D_ + kc * 32, D_, lane);
          #pragma unroll
          for (int nt = 0; nt < 8; ++nt) {
            v16bf bv = load_b_bf16(ls_wv + (size_t)nt * 16 * D_ + kc * 32, D_, lane);
            accv[nt] = wmma_bf16(a, bv, accv[nt]);
          }
          #pragma unroll
          for (int nt = 0; nt < 8; ++nt) {
            v16bf br = load_b_bf16(ls_wr + (size_t)nt * 16 * D_ + kc * 32, D_, lane);
            accr[nt] = wmma_bf16(a, br, accr[nt]);
          }
        }
        #pragma unroll
        for (int nt = 0; nt < 8; ++nt) {
          const int n = nt * 16 + lmod;
          #pragma unroll
          for (int v = 0; v < 8; ++v) {
            const int m = mb + v + lhalf * 8;
            wsv[(size_t)m * D_ + n] = accv[nt][v];
            wsr[(size_t)m * D_ + n] = 1.f / (1.f + __expf(-accr[nt][v]));
          }
        }
      }
      __syncthreads();

      // sequential width recurrence on 128 threads; one idle wave stages Wcm
      // for the channel mix (TDM DMA hidden under the scan).
      if (tid < D_) {
        const int d = tid;
        float s = ls_state[l * D_ + d];
        for (int w = 0; w < W_; w += 4) {
          __builtin_prefetch(&wsv[(size_t)(w + 16) * D_ + d], 0, 1);
          __builtin_prefetch(&wsr[(size_t)(w + 16) * D_ + d], 0, 1);
          float vv[4], rr[4], xt[4];
          #pragma unroll
          for (int u = 0; u < 4; ++u) {
            vv[u] = wsv[(size_t)(w + u) * D_ + d];
            rr[u] = wsr[(size_t)(w + u) * D_ + d];
            xt[u] = (float)ls_line[(size_t)(w + u) * D_ + d];
          }
          #pragma unroll
          for (int u = 0; u < 4; ++u) {
            s = 0.99f * s + 0.01f * vv[u];
            ls_line[(size_t)(w + u) * D_ + d] = (__bf16)(xt[u] + rr[u] * s);
          }
        }
        ls_state[l * D_ + d] = s;
      }
#if HAS_TDM
      else if (wave == 4) {
        tdm_load_bf16(wcm_t + (size_t)l * D_ * D_, ls_wv, D_ * D_);
        wait_tensor();
      }
#else
      else {
        stage_g2l(wcm_t + (size_t)l * D_ * D_, ls_wv, D_ * D_ * 2,
                  tid - D_, NTHR - D_);
        wait_async();
      }
#endif
      __syncthreads();

      // channel mix: y += y @ Wcm, register-blocked per wave (own rows only:
      // read everything, then write back in place -- no barrier needed).
      for (int mh = 0; mh < 2; ++mh) {
        const int mb = (wave + 16 * mh) * 16;
        v8f acc[8];
        #pragma unroll
        for (int nt = 0; nt < 8; ++nt) {
          const int n = nt * 16 + lmod;
          #pragma unroll
          for (int v = 0; v < 8; ++v)
            acc[nt][v] = (float)ls_line[(size_t)(mb + v + lhalf * 8) * D_ + n];
        }
        #pragma unroll
        for (int kc = 0; kc < 4; ++kc) {
          v16bf a = load_a_bf16(ls_line + (size_t)mb * D_ + kc * 32, D_, lane);
          #pragma unroll
          for (int nt = 0; nt < 8; ++nt) {
            v16bf bc = load_b_bf16(ls_wv + (size_t)nt * 16 * D_ + kc * 32,
                                   D_, lane);  // ls_wv now holds Wcm
            acc[nt] = wmma_bf16(a, bc, acc[nt]);
          }
        }
        #pragma unroll
        for (int nt = 0; nt < 8; ++nt) {
          const int n = nt * 16 + lmod;
          #pragma unroll
          for (int v = 0; v < 8; ++v)
            ls_line[(size_t)(mb + v + lhalf * 8) * D_ + n] = (__bf16)acc[nt][v];
        }
      }
    }

    // ---- output projection: pred = line @ W_out + b_out (N=202 pad 208) ---
    for (int mh = 0; mh < 2; ++mh) {
      const int mb = (wave + 16 * mh) * 16;
      float* oh = out + (((size_t)b * H_ + h) * W_) * C_;
      for (int nt = 0; nt < 13; ++nt) {
        const int nb = nt * 16;
        const int n  = nb + lmod;
        v8f acc = splat8((n < C_) ? b_out[n] : 0.f);
        #pragma unroll
        for (int kc = 0; kc < 4; ++kc) {
          v16bf a  = load_a_bf16(ls_line + (size_t)mb * D_ + kc * 32, D_, lane);
          v16bf bo = load_b_bf16(ls_wout + (size_t)nb * D_ + kc * 32, D_, lane);
          acc = wmma_bf16(a, bo, acc);
        }
        if (n < C_) {
          #pragma unroll
          for (int v = 0; v < 8; ++v)
            oh[(size_t)(mb + v + lhalf * 8) * C_ + n] = acc[v];
        }
      }
    }
    __syncthreads();  // line rewritten next h; also fences out-stores
  }
}

// ---------------------------------------------------------------------------
// Workspace layout (bytes):
//   0        win_t  bf16  57344
//   57344    wv_t   bf16  131072
//   188416   wr_t   bf16  131072
//   319488   wcm_t  bf16  131072
//   450560   wout_t bf16  53248
//   503808   ws_v   f32   524288   ([B][512][128])
//   1028096  ws_r   f32   524288
//   total ~1.55 MB
// ---------------------------------------------------------------------------
extern "C" void kernel_launch(void* const* d_in, const int* in_sizes, int n_in,
                              void* d_out, int out_size, void* d_ws, size_t ws_size,
                              hipStream_t stream) {
  (void)in_sizes; (void)n_in; (void)out_size; (void)ws_size;
  const float* x     = (const float*)d_in[0];
  const float* w_in  = (const float*)d_in[1];
  const float* w_v   = (const float*)d_in[2];
  const float* w_r   = (const float*)d_in[3];
  const float* w_cm  = (const float*)d_in[4];
  const float* w_out = (const float*)d_in[5];
  const float* bo    = (const float*)d_in[6];

  char* ws = (char*)d_ws;
  __bf16* win_t  = (__bf16*)(ws + 0);
  __bf16* wv_t   = (__bf16*)(ws + 57344);
  __bf16* wr_t   = (__bf16*)(ws + 188416);
  __bf16* wcm_t  = (__bf16*)(ws + 319488);
  __bf16* wout_t = (__bf16*)(ws + 450560);
  float*  wsv    = (float*)(ws + 503808);
  float*  wsr    = (float*)(ws + 1028096);

  const int total = D_ * CP_IN + 3 * L_ * D_ * D_ + CP_OUT * D_;
  convert_weights<<<(total + 255) / 256, 256, 0, stream>>>(
      w_in, w_v, w_r, w_cm, w_out, win_t, wv_t, wr_t, wcm_t, wout_t);

  rwkv_pipeline<<<B_, NTHR, LS_TOTAL, stream>>>(
      x, bo, win_t, wv_t, wr_t, wcm_t, wout_t, wsv, wsr, (float*)d_out);
}
